// CoreDiffusion_89601607729639
// MI455X (gfx1250) — compile-verified, matrix operand order checked by reading, not run on hardware
//
#include <hip/hip_runtime.h>

typedef __attribute__((ext_vector_type(16))) _Float16 v16h;
typedef __attribute__((ext_vector_type(8)))  _Float16 v8h;
typedef __attribute__((ext_vector_type(8)))  float    v8f;
typedef __attribute__((ext_vector_type(4)))  float    v4f;

#define N_NODES 50000
#define NPAD    50016          // padded to multiple of 32 for 32-row GEMM tiles
#define D 128
#define K_ADJ 4
#define E_EDGES 600000
#define LN_EPS 1e-5f
#define LDS_STRIDE 136         // 128 + 8 halfs pad -> 4-bank step per row

// D = A*B + C, A/B f16 16x32 / 32x16, C/D f32 16x16
#define WMMA_F16(A, B, C) \
  __builtin_amdgcn_wmma_f32_16x16x32_f16(false, (A), false, (B), (short)0, (C), false, false)

// ---------------- weight conversion (f32 -> f16, once) ----------------
__global__ void cd_cvt_weights(const float* __restrict__ wx, const float* __restrict__ wh,
                               _Float16* __restrict__ wx16, _Float16* __restrict__ wh16) {
  int i = blockIdx.x * 256 + threadIdx.x;
  if (i < 3 * D * D) {
    wx16[i] = (_Float16)wx[i];
    wh16[i] = (_Float16)wh[i];
  }
}

// ---------------- zero fill ----------------
__global__ void cd_zero(float* __restrict__ p, int n) {
  int i = blockIdx.x * 256 + threadIdx.x;
  if (i < n) p[i] = 0.0f;
}

// ---------------- SpMM scatter: res += v * x[c]  (one wave per edge) ----------------
__global__ void cd_spmm(const int* __restrict__ rows, const int* __restrict__ cols,
                        const float* __restrict__ vals, const float* __restrict__ x,
                        float* __restrict__ res) {
  int wid  = (blockIdx.x * blockDim.x + threadIdx.x) >> 5;  // edge id
  int lane = threadIdx.x & 31;
  if (wid >= E_EDGES) return;
  int   r = rows[wid];
  int   c = cols[wid];
  float v = vals[wid];
  float4 xv = *reinterpret_cast<const float4*>(x + (size_t)c * D + lane * 4);
  float* rp = res + (size_t)r * D + lane * 4;
  unsafeAtomicAdd(rp + 0, v * xv.x);
  unsafeAtomicAdd(rp + 1, v * xv.y);
  unsafeAtomicAdd(rp + 2, v * xv.z);
  unsafeAtomicAdd(rp + 3, v * xv.w);
}

// ---------------- fragment loaders (CDNA5 16-bit A/B VGPR layout) ----------------
// element e of v16h: k = kbase + ((e&8)<<1) + (e&7); kbase = k0 + 8*(lane>=16)
__device__ __forceinline__ v16h cd_ld_frag_f16g(const _Float16* __restrict__ w, int row, int kbase) {
  const v8h* p = reinterpret_cast<const v8h*>(w + (size_t)row * D + kbase);
  v8h c0 = p[0];   // k = kbase .. kbase+7
  v8h c1 = p[2];   // k = kbase+16 .. kbase+23
  v16h f;
#pragma unroll
  for (int i = 0; i < 8; ++i) {
    f[i]     = c0[i];
    f[8 + i] = c1[i];
  }
  return f;
}

__device__ __forceinline__ v16h cd_ld_frag_lds(const _Float16* s, int row, int kbase) {
  const _Float16* p = s + row * LDS_STRIDE + kbase;
  v8h c0 = *reinterpret_cast<const v8h*>(p);        // ds_load_b128
  v8h c1 = *reinterpret_cast<const v8h*>(p + 16);   // ds_load_b128
  v16h f;
#pragma unroll
  for (int i = 0; i < 8; ++i) {
    f[i]     = c0[i];
    f[8 + i] = c1[i];
  }
  return f;
}

// ---------------- fused GRU step ----------------
// block = 256 thr = 8 waves; block owns 32 rows x 128 cols of h.
// A tiles (relu(res), h) staged once in LDS as f16; wave w owns cols [16w,16w+16),
// computes 2 row-halves -> 12 WMMA accumulators, 12 WMMAs per k-step, 48 total.
__global__ __launch_bounds__(256) void cd_gru_step(
    const float* __restrict__ res, const float* __restrict__ hin, float* __restrict__ hout,
    const _Float16* __restrict__ wx16, const _Float16* __restrict__ wh16,
    const float* __restrict__ bx, const float* __restrict__ bh) {
  __shared__ _Float16 s_res[32 * LDS_STRIDE];
  __shared__ _Float16 s_h[32 * LDS_STRIDE];

  int tid  = threadIdx.x;
  int wave = tid >> 5;
  int lane = tid & 31;
  int m0   = blockIdx.x * 32;
  int n0   = wave * 16;
  int hv   = lane >> 4;      // half-wave select
  int lm   = lane & 15;
  int koff = hv * 8;

  // ---- stage A tiles to LDS (f32 -> f16, relu fused for res), once per block ----
  {
    int srow = tid >> 3;             // 0..31
    int scol = (tid & 7) * 16;       // 0,16,...,112
    const float* pr = res + (size_t)(m0 + srow) * D + scol;
    const float* ph = hin + (size_t)(m0 + srow) * D + scol;
    _Float16* sr = s_res + srow * LDS_STRIDE + scol;
    _Float16* sh = s_h   + srow * LDS_STRIDE + scol;
#pragma unroll
    for (int i = 0; i < 16; i += 8) {
      v4f r0 = *reinterpret_cast<const v4f*>(pr + i);
      v4f r1 = *reinterpret_cast<const v4f*>(pr + i + 4);
      v4f h0 = *reinterpret_cast<const v4f*>(ph + i);
      v4f h1 = *reinterpret_cast<const v4f*>(ph + i + 4);
      v8h rv, hvv;
#pragma unroll
      for (int u = 0; u < 4; ++u) {
        rv[u]     = (_Float16)fmaxf(r0[u], 0.0f);
        rv[4 + u] = (_Float16)fmaxf(r1[u], 0.0f);
        hvv[u]     = (_Float16)h0[u];
        hvv[4 + u] = (_Float16)h1[u];
      }
      *reinterpret_cast<v8h*>(sr + i) = rv;
      *reinterpret_cast<v8h*>(sh + i) = hvv;
    }
  }
  __syncthreads();

  v8f axr[2] = {v8f{}, v8f{}}, axi[2] = {v8f{}, v8f{}}, axn[2] = {v8f{}, v8f{}};
  v8f ahr[2] = {v8f{}, v8f{}}, ahi[2] = {v8f{}, v8f{}}, ahn[2] = {v8f{}, v8f{}};

#pragma unroll
  for (int k0 = 0; k0 < D; k0 += 32) {
    int kb = k0 + koff;
    // B fragments (weights, f16, shared by both row-halves)
    v16h bxr = cd_ld_frag_f16g(wx16, 0 * D + n0 + lm, kb);
    v16h bxi = cd_ld_frag_f16g(wx16, 1 * D + n0 + lm, kb);
    v16h bxn = cd_ld_frag_f16g(wx16, 2 * D + n0 + lm, kb);
    v16h bhr = cd_ld_frag_f16g(wh16, 0 * D + n0 + lm, kb);
    v16h bhi = cd_ld_frag_f16g(wh16, 1 * D + n0 + lm, kb);
    v16h bhn = cd_ld_frag_f16g(wh16, 2 * D + n0 + lm, kb);
#pragma unroll
    for (int t = 0; t < 2; ++t) {
      v16h ar = cd_ld_frag_lds(s_res, t * 16 + lm, kb);
      v16h ah = cd_ld_frag_lds(s_h,   t * 16 + lm, kb);
      axr[t] = WMMA_F16(ar, bxr, axr[t]);
      axi[t] = WMMA_F16(ar, bxi, axi[t]);
      axn[t] = WMMA_F16(ar, bxn, axn[t]);
      ahr[t] = WMMA_F16(ah, bhr, ahr[t]);
      ahi[t] = WMMA_F16(ah, bhi, ahi[t]);
      ahn[t] = WMMA_F16(ah, bhn, ahn[t]);
    }
  }

  // ---- epilogue: lane holds col j = n0+lm, rows m0 + 16t + r + 8*hv (C/D layout) ----
  int j = n0 + lm;
  float bxr_ = bx[j], bxi_ = bx[D + j], bxn_ = bx[2 * D + j];
  float bhr_ = bh[j], bhi_ = bh[D + j], bhn_ = bh[2 * D + j];
#pragma unroll
  for (int t = 0; t < 2; ++t) {
#pragma unroll
    for (int r = 0; r < 8; ++r) {
      int row = m0 + t * 16 + r + hv * 8;
      float ir  = axr[t][r] + bxr_, hr = ahr[t][r] + bhr_;
      float ii  = axi[t][r] + bxi_, hg = ahi[t][r] + bhi_;
      float inn = axn[t][r] + bxn_, hn = ahn[t][r] + bhn_;
      float resetg = 1.0f / (1.0f + __expf(-(ir + hr)));
      float inputg = 1.0f / (1.0f + __expf(-(ii + hg)));
      float newg   = tanhf(inn + resetg * hn);
      float hold   = hin[(size_t)row * D + j];
      hout[(size_t)row * D + j] = newg + inputg * (hold - newg);
    }
  }
}

// ---------------- LayerNorm: one wave per row ----------------
__global__ __launch_bounds__(256) void cd_lnorm(const float* __restrict__ h,
                                                const float* __restrict__ g,
                                                const float* __restrict__ b,
                                                float* __restrict__ out) {
  int row  = blockIdx.x * 8 + (threadIdx.x >> 5);
  int lane = threadIdx.x & 31;
  float4 v = *reinterpret_cast<const float4*>(h + (size_t)row * D + lane * 4);
  float s  = v.x + v.y + v.z + v.w;
  float s2 = v.x * v.x + v.y * v.y + v.z * v.z + v.w * v.w;
#pragma unroll
  for (int m = 16; m >= 1; m >>= 1) {
    s  += __shfl_xor(s, m, 32);
    s2 += __shfl_xor(s2, m, 32);
  }
  float mean = s * (1.0f / D);
  float var  = s2 * (1.0f / D) - mean * mean;
  float inv  = rsqrtf(var + LN_EPS);
  float4 gv = *reinterpret_cast<const float4*>(g + lane * 4);
  float4 bv = *reinterpret_cast<const float4*>(b + lane * 4);
  float4 o;
  o.x = (v.x - mean) * inv * gv.x + bv.x;
  o.y = (v.y - mean) * inv * gv.y + bv.y;
  o.z = (v.z - mean) * inv * gv.z + bv.z;
  o.w = (v.w - mean) * inv * gv.w + bv.w;
  *reinterpret_cast<float4*>(out + (size_t)row * D + lane * 4) = o;
}

extern "C" void kernel_launch(void* const* d_in, const int* in_sizes, int n_in,
                              void* d_out, int out_size, void* d_ws, size_t ws_size,
                              hipStream_t stream) {
  const float* x    = (const float*)d_in[0];   // [N, 128]
  const float* vals = (const float*)d_in[1];   // [K, E]
  const float* w_x  = (const float*)d_in[2];   // [384, 128]
  const float* b_x  = (const float*)d_in[3];   // [384]
  const float* w_h  = (const float*)d_in[4];   // [384, 128]
  const float* b_h  = (const float*)d_in[5];   // [384]
  const float* ln_g = (const float*)d_in[6];   // [128]
  const float* ln_b = (const float*)d_in[7];   // [128]
  const int*   rows = (const int*)d_in[8];     // [K, E]
  const int*   cols = (const int*)d_in[9];     // [K, E]
  float* out = (float*)d_out;

  // workspace layout (row-padded matrices so 32-row tiles divide evenly)
  char* ws = (char*)d_ws;
  size_t matB = (size_t)NPAD * D * sizeof(float);
  float*    res  = (float*)(ws);
  float*    hA   = (float*)(ws + matB);
  float*    hB   = (float*)(ws + 2 * matB);
  _Float16* wx16 = (_Float16*)(ws + 3 * matB);
  _Float16* wh16 = wx16 + 3 * D * D;

  const int NM = NPAD * D;

  cd_cvt_weights<<<(3 * D * D + 255) / 256, 256, 0, stream>>>(w_x, w_h, wx16, wh16);
  cd_zero<<<(NM + 255) / 256, 256, 0, stream>>>(hA, NM);   // h0 = 0 (incl. pad rows)

  float* hcur = hA;
  float* hnxt = hB;
  for (int kk = K_ADJ - 1; kk >= 0; --kk) {                // adj_list reversed
    cd_zero<<<(NM + 255) / 256, 256, 0, stream>>>(res, NM);
    cd_spmm<<<E_EDGES / 8, 256, 0, stream>>>(rows + (size_t)kk * E_EDGES,
                                             cols + (size_t)kk * E_EDGES,
                                             vals + (size_t)kk * E_EDGES, x, res);
    cd_gru_step<<<NPAD / 32, 256, 0, stream>>>(res, hcur, hnxt, wx16, wh16, b_x, b_h);
    float* t = hcur; hcur = hnxt; hnxt = t;
  }

  cd_lnorm<<<N_NODES / 8, 256, 0, stream>>>(hcur, ln_g, ln_b, out);
  (void)in_sizes; (void)n_in; (void)out_size; (void)ws_size;
}
